// DynamicConv_7610682048562
// MI455X (gfx1250) — matrix operand order repaired and checked by reading, not compiled, not run on hardware
//
#include <hip/hip_runtime.h>

// ---------------- Problem constants ----------------
#define BS    32
#define CIN   64
#define COUT  64
#define Hdim  128
#define Wdim  128
#define HW    (Hdim*Wdim)          // 16384
#define HIDDEN 16
#define KEXP  4
#define KTOT  576                  // CIN * 9, k = (ky*3+kx)*64 + ci
#define TEMP_INV (1.0f/30.0f)

typedef __attribute__((ext_vector_type(16))) __bf16 v16bf;
typedef __attribute__((ext_vector_type(8)))  float  v8f;
typedef int v4i __attribute__((vector_size(16)));
typedef __attribute__((address_space(1))) v4i* gas_v4i_p;   // global AS pointer to v4i
typedef __attribute__((address_space(3))) v4i* las_v4i_p;   // LDS AS pointer to v4i

__device__ __forceinline__ unsigned short f2bf(float f) {
    union { float f; unsigned u; } v; v.f = f;
    unsigned r = v.u + 0x7FFFu + ((v.u >> 16) & 1u);   // round-to-nearest-even
    return (unsigned short)(r >> 16);
}

// ---- CDNA5 async global->LDS copy (ASYNCcnt path), with safe fallback ----
#if __has_builtin(__builtin_amdgcn_global_load_async_to_lds_b128)
#define HAVE_ASYNC_LDS 1
__device__ __forceinline__ void async_copy16(const void* g, void* l) {
    __builtin_amdgcn_global_load_async_to_lds_b128(
        (gas_v4i_p)g, (las_v4i_p)l, 0, 0);
}
#else
#define HAVE_ASYNC_LDS 0
#endif

__device__ __forceinline__ void wait_asynccnt0() {
#if __has_builtin(__builtin_amdgcn_s_wait_asynccnt)
    __builtin_amdgcn_s_wait_asynccnt(0);
#else
    asm volatile("s_wait_asynccnt 0" ::: "memory");
#endif
}

// ---------------- Kernel 1: global average pool ----------------
__global__ __launch_bounds__(256) void pool_kernel(const float* __restrict__ x,
                                                   float* __restrict__ pooled) {
    int bc = blockIdx.x;                       // 0 .. BS*CIN-1
    const float* p = x + (size_t)bc * HW;
    float s = 0.f;
    for (int i = threadIdx.x; i < HW; i += 256) s += p[i];
    __shared__ float red[256];
    red[threadIdx.x] = s;
    __syncthreads();
    for (int o = 128; o > 0; o >>= 1) {
        if (threadIdx.x < o) red[threadIdx.x] += red[threadIdx.x + o];
        __syncthreads();
    }
    if (threadIdx.x == 0) pooled[bc] = red[0] * (1.0f / HW);
}

// ---------------- Kernel 2: attention MLP + softmax + agg bias ----------------
__global__ __launch_bounds__(64) void attn_kernel(const float* __restrict__ pooled,
                                                  const float* __restrict__ w1,   // [16,64]
                                                  const float* __restrict__ w2,   // [4,16]
                                                  const float* __restrict__ bias, // [4,64]
                                                  float* __restrict__ att,        // [32,4]
                                                  float* __restrict__ aggb) {     // [32,64]
    int b = threadIdx.x;
    if (b >= BS) return;
    float h[HIDDEN];
    for (int j = 0; j < HIDDEN; ++j) {
        float s = 0.f;
        for (int c = 0; c < CIN; ++c) s += pooled[b*CIN + c] * w1[j*CIN + c];
        h[j] = s > 0.f ? s : 0.f;
    }
    float lg[KEXP]; float mx = -1e30f;
    for (int k = 0; k < KEXP; ++k) {
        float s = 0.f;
        for (int j = 0; j < HIDDEN; ++j) s += h[j] * w2[k*HIDDEN + j];
        lg[k] = s * TEMP_INV;
        mx = fmaxf(mx, lg[k]);
    }
    float den = 0.f;
    for (int k = 0; k < KEXP; ++k) { lg[k] = expf(lg[k] - mx); den += lg[k]; }
    float inv = 1.0f / den;
    for (int k = 0; k < KEXP; ++k) { lg[k] *= inv; att[b*KEXP + k] = lg[k]; }
    for (int co = 0; co < COUT; ++co) {
        float s = 0.f;
        for (int k = 0; k < KEXP; ++k) s += lg[k] * bias[k*COUT + co];
        aggb[b*COUT + co] = s;
    }
}

// ---------------- Kernel 3: aggregate expert weights -> bf16, GEMM-friendly layout ----------------
// wb[b][co][k] with k = (ky*3+kx)*64 + ci   (K-contiguous per co row)
__global__ __launch_bounds__(256) void aggw_kernel(const float* __restrict__ att,
                                                   const float* __restrict__ weight, // [4,64,64,3,3]
                                                   unsigned short* __restrict__ wb) {
    int idx = blockIdx.x * 256 + threadIdx.x;     // exactly BS*COUT*KTOT threads
    int b   = idx / (COUT*KTOT);
    int rem = idx % (COUT*KTOT);
    int co  = rem / KTOT;
    int k   = rem % KTOT;
    int tap = k >> 6;            // ky*3+kx
    int ci  = k & 63;
    const float* a = att + b*KEXP;
    size_t base = ((size_t)co*CIN + ci)*9 + tap;  // within one expert
    float s = 0.f;
    #pragma unroll
    for (int k4 = 0; k4 < KEXP; ++k4)
        s += a[k4] * weight[(size_t)k4 * (COUT*CIN*9) + base];
    wb[idx] = f2bf(s);
}

// ---------------- Kernel 4: NCHW f32 -> NHWC bf16 transpose ----------------
// xt[b][h][w][c], 128 B per pixel; after this pass conv touches only xt (64 MB, L2-resident)
__global__ __launch_bounds__(256) void nhwc_kernel(const float* __restrict__ x,
                                                   unsigned short* __restrict__ xt) {
    __shared__ unsigned short tile[64 * 80];     // 64 pixels x 64 ch, row stride 80 (160 B, 16B-aligned)
    const int b  = blockIdx.x >> 8;              // 32 samples x 256 pixel-tiles
    const int p0 = (blockIdx.x & 255) * 64;
    const float* xb = x + (size_t)b * CIN * HW;
    for (int idx = threadIdx.x; idx < 64*64; idx += 256) {
        int c = idx >> 6;        // channel
        int i = idx & 63;        // pixel within tile (fast -> coalesced global reads)
        tile[i*80 + c] = f2bf(xb[(size_t)c*HW + p0 + i]);
    }
    __syncthreads();
    for (int idx = threadIdx.x; idx < 512; idx += 256) {
        int i = idx >> 3, part = idx & 7;
        *(uint4*)&xt[((size_t)b*HW + p0 + i)*64 + part*8] = *(const uint4*)&tile[i*80 + part*8];
    }
}

// ---------------- Kernel 5: dynamic conv as implicit GEMM with bf16 WMMA ----------------
// grid: BS*Hdim workgroups; one WG = one (sample, output row). 8 waves:
//   wave w: m = w&3 (co tile of 16), nb = w>>2 (pixel half of 64)
__global__ __launch_bounds__(256) void conv_kernel(const unsigned short* __restrict__ xt,
                                                   const unsigned short* __restrict__ wb,
                                                   const float* __restrict__ aggb,
                                                   float* __restrict__ out) {
    __shared__ unsigned short xs[3 * 130 * 64];   // [row 0..2][paddedCol 0..129][ci], 49920 B

    const int b = blockIdx.x >> 7;
    const int h = blockIdx.x & 127;
    const int tid = threadIdx.x;

    // zero the two halo columns (col 0 and col 129) for all 3 rows
    if (tid < 48) {
        int row = tid / 16;
        int r   = tid % 16;
        int colSel = r >> 3;            // 0 -> col 0, 1 -> col 129
        int c0 = (r & 7) * 8;
        *(uint4*)&xs[(row*130 + (colSel ? 129 : 0))*64 + c0] = make_uint4(0u,0u,0u,0u);
    }

    // stage interior: 3 rows x 128 cols x 128 B/pixel = 3072 16-byte chunks.
    // Valid rows go through the ASYNCcnt global->LDS copy engine; OOB rows are zeroed in LDS.
    for (int c = tid; c < 3072; c += 256) {
        const int hr   = c >> 10;        // 0..2  (1024 chunks per row)
        const int rem  = c & 1023;
        const int col  = rem >> 3;       // 0..127
        const int part = rem & 7;        // 16B chunk within the 128B pixel
        const int gh   = h + hr - 1;
        unsigned short* ldst = &xs[(hr*130 + col + 1)*64 + part*8];
        if (gh >= 0 && gh < Hdim) {
            const unsigned short* gsrc =
                xt + (((size_t)b*HW + (size_t)gh*Wdim + col)*64 + part*8);
#if HAVE_ASYNC_LDS
            async_copy16(gsrc, ldst);
#else
            *(uint4*)ldst = *(const uint4*)gsrc;
#endif
        } else {
            *(uint4*)ldst = make_uint4(0u,0u,0u,0u);
        }
    }
#if HAVE_ASYNC_LDS
    wait_asynccnt0();
#endif
    __syncthreads();

    const int wave = tid >> 5;
    const int lane = tid & 31;
    const int m    = wave & 3;          // co tile
    const int nb   = wave >> 2;         // pixel half (0 -> cols 0..63, 1 -> 64..127)
    const int lrow = lane & 15;
    const int half = lane >> 4;

    v8f acc[4];
    #pragma unroll
    for (int i = 0; i < 4; ++i) acc[i] = (v8f){0.f,0.f,0.f,0.f,0.f,0.f,0.f,0.f};

    const int coA = m*16 + lrow;        // A-matrix row for this lane
    const unsigned short* wrow = wb + ((size_t)(b*COUT + coA)) * KTOT;

    #pragma unroll
    for (int kk = 0; kk < 18; ++kk) {   // K = 576 in steps of 32
        const int tap    = kk >> 1;
        const int cibase = (kk & 1) * 32;
        const int ky = tap / 3;
        const int kx = tap - ky*3;

        // A fragment: contiguous k order used here (matches B) — per lane 16 contiguous K values:
        //   K = half*16 + j (j=0..15)
        union { uint4 u[2]; v16bf v; } af;
        const uint4* ap = (const uint4*)(wrow + tap*64 + cibase + half*16);
        af.u[0] = ap[0];
        af.u[1] = ap[1];                // +8 ushorts

        #pragma unroll
        for (int nt = 0; nt < 4; ++nt) {
            const int col = nb*64 + nt*16 + lrow;   // output pixel (N) for this lane
            // B fragment: same contiguous k order as A — lane group selects K half,
            //   16 contiguous K values per lane: K = half*16 + j
            union { uint4 u[2]; v16bf v; } bfm;
            const uint4* bp = (const uint4*)&xs[((ky*130) + col + kx)*64 + cibase + half*16];
            bfm.u[0] = bp[0];
            bfm.u[1] = bp[1];
            acc[nt] = __builtin_amdgcn_wmma_f32_16x16x32_bf16(
                false, af.v, false, bfm.v, (short)0, acc[nt], false, false);
        }
    }

    // store: C/D 16x16 f32 layout — VGPR j -> M = half*8 + j, N = lane%16.
    // Non-temporal: y is write-once (134 MB); keep it out of L2 so xt stays resident.
    #pragma unroll
    for (int nt = 0; nt < 4; ++nt) {
        const int w = nb*64 + nt*16 + lrow;
        #pragma unroll
        for (int j = 0; j < 8; ++j) {
            const int co = m*16 + half*8 + j;
            __builtin_nontemporal_store(acc[nt][j] + aggb[b*COUT + co],
                                        &out[(((size_t)b*COUT + co)*Hdim + h)*Wdim + w]);
        }
    }
}

// ---------------- Host launcher ----------------
extern "C" void kernel_launch(void* const* d_in, const int* in_sizes, int n_in,
                              void* d_out, int out_size, void* d_ws, size_t ws_size,
                              hipStream_t stream) {
    const float* x   = (const float*)d_in[0];   // [32,64,128,128]
    const float* w1  = (const float*)d_in[1];   // [16,64]
    const float* w2  = (const float*)d_in[2];   // [4,16]
    const float* wgt = (const float*)d_in[3];   // [4,64,64,3,3]
    const float* bia = (const float*)d_in[4];   // [4,64]
    float* out = (float*)d_out;

    // workspace layout
    char* ws = (char*)d_ws;
    float* pooled = (float*)(ws);                       // 2048 f32  = 8192 B
    float* att    = (float*)(ws + 8192);                // 128  f32  = 512 B
    float* aggb   = (float*)(ws + 8704);                // 2048 f32  = 8192 B
    unsigned short* wb = (unsigned short*)(ws + 16896); // 32*64*576 bf16 = 2359296 B
    unsigned short* xt = (unsigned short*)(ws + 2376192); // NHWC bf16: 32*16384*64*2 = 64 MB

    pool_kernel<<<dim3(BS*CIN), dim3(256), 0, stream>>>(x, pooled);
    attn_kernel<<<dim3(1), dim3(64), 0, stream>>>(pooled, w1, w2, bia, att, aggb);
    aggw_kernel<<<dim3((BS*COUT*KTOT)/256), dim3(256), 0, stream>>>(att, wgt, wb);
    nhwc_kernel<<<dim3(BS*256), dim3(256), 0, stream>>>(x, xt);
    conv_kernel<<<dim3(BS*Hdim), dim3(256), 0, stream>>>(xt, wb, aggb, out);
}
